// HVAE_89764816487077
// MI455X (gfx1250) — compile-verified, matrix operand order checked by reading, not run on hardware
//
#include <hip/hip_runtime.h>
#include <hip/hip_bf16.h>

typedef __attribute__((ext_vector_type(16))) __bf16 v16bf;
typedef __attribute__((ext_vector_type(8)))  float  v8f;

#define DEV __device__ __forceinline__
static constexpr int BN = 2048;

// padded strides
static constexpr int LD20   = 64;    // 20-dim vectors
static constexpr int LD450  = 512;   // 450-dim activations
static constexpr int LD1568 = 1600;  // 1568-dim activations (7x7x32 conv layout)
static constexpr int LD3136 = 6272;  // 14x14 x 32-channel-slot conv layout

DEV float sp_f(float x)  { return (x > 20.f) ? x : log1pf(__expf(x)); }
DEV float sig_f(float x) { return 1.f / (1.f + __expf(-x)); }

DEV v8f wmma_bf16(v16bf a, v16bf b, v8f c) {
  return __builtin_amdgcn_wmma_f32_16x16x32_bf16(false, a, false, b, (short)0, c,
                                                 false, false);
}

// A fragment 16x32 (ISA 7.12.2): lane 0-15 row M=lane, halves 0..7 = K kb..kb+7,
// halves 8..15 = K kb+16..kb+23; lanes 16-31 same rows, kb += 8.
// Operand region is padded+zeroed => unconditional contiguous vector loads.
DEV v16bf cvt16(float4 f0, float4 f1, float4 f2, float4 f3) {
  v16bf a;
  a[0]=(__bf16)f0.x;  a[1]=(__bf16)f0.y;  a[2]=(__bf16)f0.z;  a[3]=(__bf16)f0.w;
  a[4]=(__bf16)f1.x;  a[5]=(__bf16)f1.y;  a[6]=(__bf16)f1.z;  a[7]=(__bf16)f1.w;
  a[8]=(__bf16)f2.x;  a[9]=(__bf16)f2.y;  a[10]=(__bf16)f2.z; a[11]=(__bf16)f2.w;
  a[12]=(__bf16)f3.x; a[13]=(__bf16)f3.y; a[14]=(__bf16)f3.z; a[15]=(__bf16)f3.w;
  return a;
}
DEV v16bf load_a16(const float* __restrict__ rowBase, int k0, int lane) {
  const float* p = rowBase + k0 + ((lane & 16) ? 8 : 0);
  const float4* q  = (const float4*)p;
  const float4* q2 = (const float4*)(p + 16);
  return cvt16(q[0], q[1], q2[0], q2[1]);
}

// ---- FC GEMM: C[2048,N] = A[2048,K] x W, W pre-swizzled: tiles [nt][kt][lane][16]
// EPI: 0=raw, 1=bias+softplus(post), 2=bias+softplus(pre&post), 3=bias raw
template<int EPI>
__global__ __launch_bounds__(256) void fc_gemm(
    const float* __restrict__ A, int lda,
    const __bf16* __restrict__ W, int Ktiles,
    const float* __restrict__ bias,
    float* __restrict__ out0, float* __restrict__ out1,
    int N, int ldc)
{
  int wave = threadIdx.x >> 5, lane = threadIdx.x & 31;
  int m0 = (blockIdx.x * 8 + wave) * 16;
  int n0 = blockIdx.y * 64;
  v8f acc0 = {}, acc1 = {}, acc2 = {}, acc3 = {};
  const float* Arow = A + (size_t)(m0 + (lane & 15)) * lda;
  size_t tstep = (size_t)Ktiles * 512;
  const __bf16* Wl = W + (size_t)(blockIdx.y * 4) * tstep + lane * 16;
  for (int kt = 0; kt < Ktiles; ++kt) {
    __builtin_prefetch(Arow + kt * 32 + 256, 0, 0);
    v16bf a  = load_a16(Arow, kt * 32, lane);
    const __bf16* wp = Wl + (size_t)kt * 512;
    v16bf b0 = *(const v16bf*)(wp);
    v16bf b1 = *(const v16bf*)(wp + tstep);
    v16bf b2 = *(const v16bf*)(wp + 2 * tstep);
    v16bf b3 = *(const v16bf*)(wp + 3 * tstep);
    acc0 = wmma_bf16(a, b0, acc0);
    acc1 = wmma_bf16(a, b1, acc1);
    acc2 = wmma_bf16(a, b2, acc2);
    acc3 = wmma_bf16(a, b3, acc3);
  }
  int rbase = m0 + ((lane & 16) ? 8 : 0);
  v8f accs[4] = {acc0, acc1, acc2, acc3};
#pragma unroll
  for (int j = 0; j < 4; ++j) {
    int col = n0 + j * 16 + (lane & 15);
    bool ok = col < N;
    float bv = (EPI != 0 && ok) ? bias[col] : 0.f;
#pragma unroll
    for (int r = 0; r < 8; ++r) {
      float v = ok ? (accs[j][r] + bv) : 0.f;   // pads stored as zero
      size_t o = (size_t)(rbase + r) * ldc + col;
      if      (EPI == 2) { out0[o] = v; out1[o] = ok ? sp_f(v) : 0.f; }
      else if (EPI == 1) { out1[o] = ok ? sp_f(v) : 0.f; }
      else               { out0[o] = v; }
    }
  }
}

// ---- Tap-GEMM conv. Activations: [b][pixel][32-channel slot].
// Wt pre-swizzled: [25 taps][CoutTiles][lane][16], K padded to 32 (zeros).
// MODE 0 (conv, stride s):  iy = oy*s + dy - 2
// MODE 1 (convT, stride s): t = oy + dy - 2 valid iff t>=0, t%s==0, t/s<inH
template<int EPI, int MODE>
__global__ __launch_bounds__(256) void conv_tap(
    const float* __restrict__ In, int ldin,
    const __bf16* __restrict__ Wt,
    const float* __restrict__ bias,
    float* __restrict__ out0, float* __restrict__ out1, int ldout,
    int Cout, int inH, int outH, int stride)
{
  int wave = threadIdx.x >> 5, lane = threadIdx.x & 31;
  int m0 = (blockIdx.x * 8 + wave) * 16;
  int po = blockIdx.y;                  // wave-uniform tap control flow
  int oy = po / outH, ox = po % outH;
  int coutTiles = Cout >> 4;
  v8f acc0 = {}, acc1 = {};
  const float* base = In + (size_t)(m0 + (lane & 15)) * ldin;
  for (int dy = 0; dy < 5; ++dy) {
    int iy;
    if (MODE == 0) { iy = oy * stride + dy - 2; if (iy < 0 || iy >= inH) continue; }
    else { int t = oy + dy - 2; if (t < 0 || (t % stride) || (t / stride) >= inH) continue; iy = t / stride; }
    for (int dx = 0; dx < 5; ++dx) {
      int ix;
      if (MODE == 0) { ix = ox * stride + dx - 2; if (ix < 0 || ix >= inH) continue; }
      else { int t = ox + dx - 2; if (t < 0 || (t % stride) || (t / stride) >= inH) continue; ix = t / stride; }
      int pi = iy * inH + ix;
      v16bf a = load_a16(base, pi * 32, lane);
      const __bf16* wp = Wt + (size_t)(dy * 5 + dx) * coutTiles * 512 + lane * 16;
      v16bf b0 = *(const v16bf*)wp;
      acc0 = wmma_bf16(a, b0, acc0);
      if (Cout > 16) {
        v16bf b1 = *(const v16bf*)(wp + 512);
        acc1 = wmma_bf16(a, b1, acc1);
      }
    }
  }
  int rbase = m0 + ((lane & 16) ? 8 : 0);
  v8f accs[2] = {acc0, acc1};
  for (int j = 0; j < coutTiles; ++j) {
    int co = j * 16 + (lane & 15);
    float bv = (EPI == 0) ? 0.f : bias[co];
#pragma unroll
    for (int r = 0; r < 8; ++r) {
      float v = accs[j][r] + bv;
      size_t o = (size_t)(rbase + r) * ldout + (size_t)po * 32 + co;
      if      (EPI == 2) { out0[o] = v; out1[o] = sp_f(v); }
      else if (EPI == 1) { out1[o] = sp_f(v); }
      else               { out0[o] = v; }
    }
  }
}

// ---- VALU kernels (WMMA-hostile shapes: N=1 / K=1 / K=25) ----
__global__ __launch_bounds__(256) void conv1_k(const float* __restrict__ x,
    const float* __restrict__ w, const float* __restrict__ b, float* __restrict__ out)
{
  int idx = blockIdx.x * 256 + threadIdx.x;
  if (idx >= BN * 16 * 196) return;
  int po = idx % 196, co = (idx / 196) % 16, bn = idx / (196 * 16);
  int oy = po / 14, ox = po % 14;
  float acc = b[co];
  const float* xs = x + (size_t)bn * 784;
  const float* ws = w + co * 25;
#pragma unroll
  for (int dy = 0; dy < 5; ++dy) {
    int iy = 2 * oy + dy - 2; if (iy < 0 || iy >= 28) continue;
#pragma unroll
    for (int dx = 0; dx < 5; ++dx) {
      int ix = 2 * ox + dx - 2; if (ix < 0 || ix >= 28) continue;
      acc += ws[dy * 5 + dx] * xs[iy * 28 + ix];
    }
  }
  out[(size_t)bn * LD3136 + (size_t)po * 32 + co] = sp_f(acc);
}

__global__ __launch_bounds__(256) void d3_fwd_k(const float* __restrict__ h4,
    const float* __restrict__ w, const float* __restrict__ b, float* __restrict__ logits)
{
  int idx = blockIdx.x * 256 + threadIdx.x;
  if (idx >= BN * 784) return;
  int po = idx % 784, bn = idx / 784;
  int oy = po / 28, ox = po % 28;
  float acc = b[0];
  const float* hs = h4 + (size_t)bn * LD3136;
  for (int dy = 0; dy < 5; ++dy) {
    int ty = oy + dy - 2; if (ty < 0 || (ty & 1) || (ty >> 1) >= 14) continue;
    int iy = ty >> 1;
    for (int dx = 0; dx < 5; ++dx) {
      int tx = ox + dx - 2; if (tx < 0 || (tx & 1) || (tx >> 1) >= 14) continue;
      int ix = tx >> 1;
      int wb = (4 - dy) * 5 + (4 - dx);
      const float* hp = hs + (iy * 14 + ix) * 32;
#pragma unroll
      for (int ci = 0; ci < 16; ++ci)
        acc += w[ci * 25 + wb] * hp[ci];
    }
  }
  logits[(size_t)bn * 784 + po] = acc;
}

__global__ __launch_bounds__(256) void d3_bwd_k(const float* __restrict__ dlog,
    const float* __restrict__ w, float* __restrict__ dh4)
{
  int idx = blockIdx.x * 256 + threadIdx.x;
  if (idx >= BN * 16 * 196) return;
  int pi = idx % 196, ci = (idx / 196) % 16, bn = idx / (196 * 16);
  int iy = pi / 14, ix = pi % 14;
  float acc = 0.f;
  const float* ds = dlog + (size_t)bn * 784;
  const float* ws = w + ci * 25;
#pragma unroll
  for (int dy = 0; dy < 5; ++dy) {
    int oy = 2 * iy + dy - 2; if (oy < 0 || oy >= 28) continue;
#pragma unroll
    for (int dx = 0; dx < 5; ++dx) {
      int ox = 2 * ix + dx - 2; if (ox < 0 || ox >= 28) continue;
      acc += ws[dy * 5 + dx] * ds[oy * 28 + ox];
    }
  }
  dh4[(size_t)bn * LD3136 + (size_t)pi * 32 + ci] = acc;
}

// ---- glue ----
__global__ void fill0_k(float* __restrict__ p, unsigned n) {
  unsigned i = blockIdx.x * 256 + threadIdx.x;
  if (i < n) p[i] = 0.f;
}

__global__ void scalars_k(const float* lfr, const float* t0r, float* scal) {
  int t = threadIdx.x;
  float T0 = 1.f + __expf(t0r[0]);
  if (t < 20) scal[t] = sig_f(lfr[t]) * 0.5f;
  if (t == 20) scal[40] = sqrtf(T0);
  if (t < 5) {
    float kv = (float)(t + 1), km = kv - 1.f;
    float ts  = (1.f - T0) * kv * kv / 25.f + T0;
    float tsm = (1.f - T0) * km * km / 25.f + T0;
    scal[32 + t] = sqrtf(ts / tsm);
  }
}

// Pack FC weight into swizzled B-fragment tiles: dst[nt][kt][lane][16].
// logical W(k,n); perm maps storage index s -> original feature (s%32)*49 + s/32.
__global__ void pack_fc_sw(const float* __restrict__ src, __bf16* __restrict__ dst,
                           int srcLd, int Kreal, int Nreal, int Ktiles, int Ntiles,
                           int kIsRow, int permK, int permN) {
  int i = blockIdx.x * 256 + threadIdx.x;
  if (i >= Ntiles * Ktiles * 512) return;
  int nt = i / (Ktiles * 512);
  int r  = i % (Ktiles * 512);
  int kt = r / 512, e = r % 512;
  int lane = e / 16, h = e % 16;
  int k = kt * 32 + ((lane & 16) ? 8 : 0) + (h < 8 ? h : 16 + (h - 8));
  int n = nt * 16 + (lane & 15);
  float v = 0.f;
  if (k < Kreal && n < Nreal) {
    int kk = permK ? (k % 32) * 49 + k / 32 : k;
    int nn = permN ? (n % 32) * 49 + n / 32 : n;
    v = kIsRow ? src[(size_t)kk * srcLd + nn] : src[(size_t)nn * srcLd + kk];
  }
  dst[i] = (__bf16)v;
}

// Pack conv weight [D0][D1][5][5] into tap tiles dst[tap][CoutTile][lane][16].
__global__ void pack_conv_sw(const float* __restrict__ src, __bf16* __restrict__ dst,
                             int D0, int D1, int transpose, int flip) {
  int Kd = transpose ? D1 : D0;
  int Nd = transpose ? D0 : D1;
  int ct = Nd >> 4;
  int i = blockIdx.x * 256 + threadIdx.x;
  if (i >= 25 * ct * 512) return;
  int tap = i / (ct * 512);
  int r   = i % (ct * 512);
  int tile = r / 512, e = r % 512;
  int lane = e / 16, h = e % 16;
  int k = ((lane & 16) ? 8 : 0) + (h < 8 ? h : 16 + (h - 8));
  int n = tile * 16 + (lane & 15);
  float v = 0.f;
  if (k < Kd && n < Nd) {
    int dy = tap / 5, dx = tap % 5;
    int ts = flip ? ((4 - dy) * 5 + (4 - dx)) : tap;
    int a0 = transpose ? n : k;
    int a1 = transpose ? k : n;
    v = src[((size_t)a0 * D1 + a1) * 25 + ts];
  }
  dst[i] = (__bf16)v;
}

__global__ void dlog_k(float* __restrict__ logits, const float* __restrict__ x, int n) {
  int i = blockIdx.x * 256 + threadIdx.x;
  if (i < n) logits[i] = sig_f(logits[i]) - x[i];
}

__global__ void gate_k(float* __restrict__ g, const float* __restrict__ pre, int n) {
  int i = blockIdx.x * 256 + threadIdx.x;
  if (i < n) g[i] *= sig_f(pre[i]);   // zero pads stay zero: 0 * sig(0)
}

__global__ void initzp_k(float* z, float* p, const float* mu, const float* sg,
                         const float* ez, const float* ep, const float* scal) {
  int i = blockIdx.x * 256 + threadIdx.x;
  if (i >= BN * 20) return;
  int o = (i / 20) * LD20 + (i % 20);
  z[o] = mu[o] + sg[o] * ez[i];
  p[o] = scal[40] * ep[i];
}

__global__ void leap1_k(float* z, const float* p, float* ph, const float* dz,
                        const float* scal) {
  int i = blockIdx.x * 256 + threadIdx.x;
  if (i >= BN * 20) return;
  int j = i % 20, o = (i / 20) * LD20 + j;
  float lf = scal[j];
  float d  = dz[o] + z[o];
  float pv = p[o] - 0.5f * lf * d;
  ph[o] = pv;
  z[o]  = z[o] + lf * pv;
}

__global__ void leap2_k(float* z, float* p, const float* ph, const float* dz,
                        const float* scal, int k) {
  int i = blockIdx.x * 256 + threadIdx.x;
  if (i >= BN * 20) return;
  int j = i % 20, o = (i / 20) * LD20 + j;
  float lf = scal[j], al = scal[32 + k];
  float d  = dz[o] + z[o];
  p[o] = al * (ph[o] - 0.5f * lf * d);
}

__global__ __launch_bounds__(256) void ell_k(const float* __restrict__ x,
    const float* __restrict__ logits, float* __restrict__ ell) {
  __shared__ float s[256];
  int bn = blockIdx.x;
  float a = 0.f;
  for (int i = threadIdx.x; i < 784; i += 256) {
    float l = logits[(size_t)bn * 784 + i];
    a += x[(size_t)bn * 784 + i] * l - sp_f(l);
  }
  s[threadIdx.x] = a; __syncthreads();
  for (int st = 128; st > 0; st >>= 1) {
    if (threadIdx.x < st) s[threadIdx.x] += s[threadIdx.x + st];
    __syncthreads();
  }
  if (threadIdx.x == 0) ell[bn] = s[0];
}

__global__ void nkl_k(const float* z, const float* p, const float* sg, float* nkl) {
  int bn = blockIdx.x * 256 + threadIdx.x;
  if (bn >= BN) return;
  float a = 20.f;
  for (int j = 0; j < 20; ++j) {
    float zv = z[bn * LD20 + j], pv = p[bn * LD20 + j];
    a += -0.5f * zv * zv - 0.5f * pv * pv + __logf(sg[bn * LD20 + j]);
  }
  nkl[bn] = a;
}

__global__ __launch_bounds__(256) void final_k(const float* ell, const float* nkl,
                                               float* out) {
  __shared__ float s[256];
  float a = 0.f;
  for (int i = threadIdx.x; i < BN; i += 256) a += ell[i] + nkl[i];
  s[threadIdx.x] = a; __syncthreads();
  for (int st = 128; st > 0; st >>= 1) {
    if (threadIdx.x < st) s[threadIdx.x] += s[threadIdx.x + st];
    __syncthreads();
  }
  if (threadIdx.x == 0) out[0] = s[0] / (float)BN;
}

extern "C" void kernel_launch(void* const* d_in, const int* in_sizes, int n_in,
                              void* d_out, int out_size, void* d_ws, size_t ws_size,
                              hipStream_t stream)
{
  (void)in_sizes; (void)n_in; (void)out_size; (void)ws_size;
  const float* x    = (const float*)d_in[0];
  const float* ez   = (const float*)d_in[1];
  const float* ep   = (const float*)d_in[2];
  const float* c1w  = (const float*)d_in[3];  const float* c1b = (const float*)d_in[4];
  const float* c2w  = (const float*)d_in[5];  const float* c2b = (const float*)d_in[6];
  const float* c3w  = (const float*)d_in[7];  const float* c3b = (const float*)d_in[8];
  const float* fc1w = (const float*)d_in[9];  const float* fc1b= (const float*)d_in[10];
  const float* fmuw = (const float*)d_in[11]; const float* fmub= (const float*)d_in[12];
  const float* fsgw = (const float*)d_in[13]; const float* fsgb= (const float*)d_in[14];
  const float* g1w  = (const float*)d_in[15]; const float* g1b = (const float*)d_in[16];
  const float* g2w  = (const float*)d_in[17]; const float* g2b = (const float*)d_in[18];
  const float* d1w  = (const float*)d_in[19]; const float* d1b = (const float*)d_in[20];
  const float* d2w  = (const float*)d_in[21]; const float* d2b = (const float*)d_in[22];
  const float* d3w  = (const float*)d_in[23]; const float* d3b = (const float*)d_in[24];
  const float* lfr  = (const float*)d_in[25]; const float* t0r = (const float*)d_in[26];

  float* wsf = (float*)d_ws;
  size_t off = 0;
  auto alloc = [&](size_t n) { size_t o = off; off += n; return o; };
  size_t o_scal = alloc(64);
  // zero-initialized region: six [B x 64] vectors + the two 32-slot 14x14 tensors
  size_t o_z  = alloc(BN * LD20), o_p  = alloc(BN * LD20), o_ph = alloc(BN * LD20);
  size_t o_dz = alloc(BN * LD20), o_mu = alloc(BN * LD20), o_sg = alloc(BN * LD20);
  size_t o_h4p = alloc((size_t)BN * LD3136), o_h4 = alloc((size_t)BN * LD3136);
  size_t zero_len = 6 * (size_t)BN * LD20 + 2 * (size_t)BN * LD3136;
  size_t o_h1p = alloc((size_t)BN * LD450),  o_h1 = alloc((size_t)BN * LD450);
  size_t o_h2p = alloc((size_t)BN * LD1568), o_h2 = alloc((size_t)BN * LD1568);
  size_t o_h3p = alloc((size_t)BN * LD1568), o_h3 = alloc((size_t)BN * LD1568);
  size_t o_log = alloc((size_t)BN * 784);
  size_t o_ell = alloc(BN), o_nkl = alloc(BN);

  __bf16* wbb = (__bf16*)(wsf + ((off + 63) & ~(size_t)63));
  size_t ho = 0;
  auto halloc = [&](size_t n) { size_t o = ho; ho += n; return o; };
  // fc packs: Ntiles*Ktiles*512
  size_t b_g1f = halloc((size_t)32 * 1 * 512);    // K=20 (1), N=450 (32 tiles)
  size_t b_g2f = halloc((size_t)100 * 15 * 512);  // K=450(15), N=1568(100)
  size_t b_fc1 = halloc((size_t)32 * 49 * 512);   // K=1568(49), N=450(32)
  size_t b_fmu = halloc((size_t)4 * 15 * 512);    // K=450(15), N=20(4)
  size_t b_fsg = halloc((size_t)4 * 15 * 512);
  size_t b_g2b = halloc((size_t)32 * 49 * 512);
  size_t b_g1b = halloc((size_t)4 * 15 * 512);
  // conv tap packs: 25 * CoutTiles * 512
  size_t b_c2  = halloc((size_t)25 * 2 * 512);
  size_t b_c3  = halloc((size_t)25 * 2 * 512);
  size_t b_d1f = halloc((size_t)25 * 2 * 512);
  size_t b_d1b = halloc((size_t)25 * 2 * 512);
  size_t b_d2f = halloc((size_t)25 * 1 * 512);
  size_t b_d2b = halloc((size_t)25 * 2 * 512);

  float *Z = wsf + o_z, *P = wsf + o_p, *PH = wsf + o_ph, *DZ = wsf + o_dz;
  float *MU = wsf + o_mu, *SG = wsf + o_sg;
  float *H1P = wsf + o_h1p, *H1 = wsf + o_h1;
  float *H2P = wsf + o_h2p, *H2 = wsf + o_h2;
  float *H3P = wsf + o_h3p, *H3 = wsf + o_h3;
  float *H4P = wsf + o_h4p, *H4 = wsf + o_h4;
  float *LOG = wsf + o_log, *SCAL = wsf + o_scal;
  float *ELL = wsf + o_ell, *NKL = wsf + o_nkl;
  float *IC1 = H4P, *IC2 = H2P, *IC3 = H3P, *HF = H1P;  // inference-time aliases

  dim3 blk(256);
  auto g1g = [](size_t n) { return dim3((unsigned)((n + 255) / 256)); };

  // zero the regions whose pads are not refreshed by padded GEMM stores
  fill0_k<<<g1g(zero_len), blk, 0, stream>>>(wsf + o_z, (unsigned)zero_len);
  scalars_k<<<dim3(1), dim3(32), 0, stream>>>(lfr, t0r, SCAL);

  // ---- pack weights (deterministic, every call) ----
  pack_fc_sw<<<g1g((size_t)32*1*512),  blk, 0, stream>>>(g1w,  wbb+b_g1f, 20,   20,  450, 1, 32, 0, 0, 0);
  pack_fc_sw<<<g1g((size_t)100*15*512),blk, 0, stream>>>(g2w,  wbb+b_g2f, 450,  450, 1568,15,100, 0, 0, 1);
  pack_fc_sw<<<g1g((size_t)32*49*512), blk, 0, stream>>>(fc1w, wbb+b_fc1, 1568, 1568,450, 49, 32, 0, 1, 0);
  pack_fc_sw<<<g1g((size_t)4*15*512),  blk, 0, stream>>>(fmuw, wbb+b_fmu, 450,  450, 20, 15, 4,  0, 0, 0);
  pack_fc_sw<<<g1g((size_t)4*15*512),  blk, 0, stream>>>(fsgw, wbb+b_fsg, 450,  450, 20, 15, 4,  0, 0, 0);
  pack_fc_sw<<<g1g((size_t)32*49*512), blk, 0, stream>>>(g2w,  wbb+b_g2b, 450,  1568,450, 49, 32, 1, 1, 0);
  pack_fc_sw<<<g1g((size_t)4*15*512),  blk, 0, stream>>>(g1w,  wbb+b_g1b, 20,   450, 20, 15, 4,  1, 0, 0);
  pack_conv_sw<<<g1g((size_t)25*2*512), blk, 0, stream>>>(c2w, wbb+b_c2, 32, 16, 1, 0);
  pack_conv_sw<<<g1g((size_t)25*2*512), blk, 0, stream>>>(c3w, wbb+b_c3, 32, 32, 1, 0);
  pack_conv_sw<<<g1g((size_t)25*2*512), blk, 0, stream>>>(d1w, wbb+b_d1f, 32, 32, 0, 1);
  pack_conv_sw<<<g1g((size_t)25*2*512), blk, 0, stream>>>(d1w, wbb+b_d1b, 32, 32, 1, 0);
  pack_conv_sw<<<g1g((size_t)25*1*512), blk, 0, stream>>>(d2w, wbb+b_d2f, 32, 16, 0, 1);
  pack_conv_sw<<<g1g((size_t)25*2*512), blk, 0, stream>>>(d2w, wbb+b_d2b, 32, 16, 1, 0);

  // ---- inference network ----
  conv1_k<<<g1g((size_t)BN*16*196), blk, 0, stream>>>(x, c1w, c1b, IC1);
  conv_tap<1,0><<<dim3(16,49), blk, 0, stream>>>(IC1, LD3136, wbb+b_c2, c2b,
                                                 nullptr, IC2, LD1568, 32, 14, 7, 2);
  conv_tap<1,0><<<dim3(16,49), blk, 0, stream>>>(IC2, LD1568, wbb+b_c3, c3b,
                                                 nullptr, IC3, LD1568, 32, 7, 7, 1);
  fc_gemm<1><<<dim3(16,8), blk, 0, stream>>>(IC3, LD1568, wbb+b_fc1, 49, fc1b,
                                             nullptr, HF, 450, LD450);
  fc_gemm<3><<<dim3(16,1), blk, 0, stream>>>(HF, LD450, wbb+b_fmu, 15, fmub,
                                             MU, nullptr, 20, LD20);
  fc_gemm<1><<<dim3(16,1), blk, 0, stream>>>(HF, LD450, wbb+b_fsg, 15, fsgb,
                                             nullptr, SG, 20, LD20);
  initzp_k<<<g1g((size_t)BN*20), blk, 0, stream>>>(Z, P, MU, SG, ez, ep, SCAL);

  auto run_gen_fwd = [&]() {
    fc_gemm<2><<<dim3(16,8), blk, 0, stream>>>(Z, LD20, wbb+b_g1f, 1, g1b,
                                               H1P, H1, 450, LD450);
    fc_gemm<2><<<dim3(16,25), blk, 0, stream>>>(H1, LD450, wbb+b_g2f, 15, g2b,
                                                H2P, H2, 1568, LD1568);
    conv_tap<2,1><<<dim3(16,49), blk, 0, stream>>>(H2, LD1568, wbb+b_d1f, d1b,
                                                   H3P, H3, LD1568, 32, 7, 7, 1);
    conv_tap<2,1><<<dim3(16,196), blk, 0, stream>>>(H3, LD1568, wbb+b_d2f, d2b,
                                                    H4P, H4, LD3136, 16, 7, 14, 2);
    d3_fwd_k<<<g1g((size_t)BN*784), blk, 0, stream>>>(H4, d3w, d3b, LOG);
  };

  auto run_dU = [&]() {
    run_gen_fwd();
    dlog_k<<<g1g((size_t)BN*784), blk, 0, stream>>>(LOG, x, BN*784);
    d3_bwd_k<<<g1g((size_t)BN*16*196), blk, 0, stream>>>(LOG, d3w, H4);
    gate_k<<<g1g((size_t)BN*LD3136), blk, 0, stream>>>(H4, H4P, BN*LD3136);
    conv_tap<0,0><<<dim3(16,49), blk, 0, stream>>>(H4, LD3136, wbb+b_d2b, nullptr,
                                                   H3, nullptr, LD1568, 32, 14, 7, 2);
    gate_k<<<g1g((size_t)BN*LD1568), blk, 0, stream>>>(H3, H3P, BN*LD1568);
    conv_tap<0,1><<<dim3(16,49), blk, 0, stream>>>(H3, LD1568, wbb+b_d1b, nullptr,
                                                   H2, nullptr, LD1568, 32, 7, 7, 1);
    gate_k<<<g1g((size_t)BN*LD1568), blk, 0, stream>>>(H2, H2P, BN*LD1568);
    fc_gemm<0><<<dim3(16,8), blk, 0, stream>>>(H2, LD1568, wbb+b_g2b, 49, nullptr,
                                               H1, nullptr, 450, LD450);
    gate_k<<<g1g((size_t)BN*LD450), blk, 0, stream>>>(H1, H1P, BN*LD450);
    fc_gemm<0><<<dim3(16,1), blk, 0, stream>>>(H1, LD450, wbb+b_g1b, 15, nullptr,
                                               DZ, nullptr, 20, LD20);
  };

  for (int k = 0; k < 5; ++k) {
    run_dU();
    leap1_k<<<g1g((size_t)BN*20), blk, 0, stream>>>(Z, P, PH, DZ, SCAL);
    run_dU();
    leap2_k<<<g1g((size_t)BN*20), blk, 0, stream>>>(Z, P, PH, DZ, SCAL, k);
  }

  run_gen_fwd();
  ell_k<<<dim3(BN), blk, 0, stream>>>(x, LOG, ELL);
  nkl_k<<<g1g(BN), blk, 0, stream>>>(Z, P, SG, NKL);
  final_k<<<dim3(1), blk, 0, stream>>>(ELL, NKL, (float*)d_out);
}
